// Model_2078764172048
// MI455X (gfx1250) — compile-verified, matrix operand order checked by reading, not run on hardware
//
#include <hip/hip_runtime.h>

// ---------- types ----------
typedef __attribute__((ext_vector_type(8)))  int      v8i;
typedef __attribute__((ext_vector_type(4)))  int      v4i;
typedef __attribute__((ext_vector_type(4)))  _Float16 v4h;
typedef __attribute__((ext_vector_type(8)))  float    v8f;
typedef __attribute__((ext_vector_type(4)))  float    v4f;

#define CH      256
#define PITB    272                 // fp8 LDS row pitch (bytes): 68 dwords -> conflict-free b64 frags
#define P0PIT   264                 // f16 residual buffer pitch (halfs)
#define MROWS   32                  // rows per wave (2 x 16-row WMMA subtiles)
#define WAVES   8
#define RPB     (MROWS * WAVES)     // 256 rows per block

// d_ws byte offsets (fp8 weights + f32 bin columns)
#define OFF_D1   0u                 // 256x256 fp8
#define OFF_D2   65536u
#define OFF_PC   131072u
#define OFF_PL   196608u            // 256x256 (row 255 zero-padded)
#define OFF_U1A  262144u            // 256x256 (cols 0..255 of w_up1)
#define OFF_R1   327680u
#define OFF_R2   393216u
#define OFF_UO   458752u            // 2048x256 fp8
#define OFF_U1B  983040u            // 256x8 f32 (cols 256..263 of w_up1)

// LDS layout
#define WCHUNK_B   (64 * PITB)                  // weight chunk: 4 ntiles = 64 rows -> 17408 B
#define LDS_SLAB0  (2 * WCHUNK_B)               // 34816 B double-buffered weight chunks
#define SLAB_P0    0u                           // 32 x P0PIT f16 = 16896 B
#define SLAB_P1    16896u                       // 32 x PITB fp8 = 8704 B
#define SLAB_P2    25600u
#define SLAB_BIN   34304u                       // 32 x 8 f32 = 1024 B
#define SLAB_B     35328u
#define LDS_TOTAL  (LDS_SLAB0 + WAVES * SLAB_B) // 317440 B

// branch-free f32 -> FP8 E4M3.
// Scale by 2^-8 so the f16 exponent field equals the e4m3 exponent field
// (denormals map exactly as well), then round m10 -> m3 and clamp to 0x7E (448).
__device__ __forceinline__ unsigned int f32_to_e4m3(float v) {
  v = fminf(fmaxf(v, -448.f), 448.f) * 0.00390625f;       // * 2^-8
  unsigned short hb = __builtin_bit_cast(unsigned short, (_Float16)v);
  unsigned int r = (((unsigned int)hb & 0x7fffu) + 0x40u) >> 7;
  r = r > 0x7eu ? 0x7eu : r;
  return (((unsigned int)hb >> 8) & 0x80u) | r;
}

// requant (clamp(v/256, [-128,127])) fused with the 2^-8 e4m3 scaling:
// requant(v) * 2^-8 == clamp(v * 2^-16, [-0.5, 127/256])
__device__ __forceinline__ unsigned int q8(float v) {
  v = v * (1.0f / 65536.0f);
  v = fminf(fmaxf(v, -0.5f), 0.49609375f);
  unsigned short hb = __builtin_bit_cast(unsigned short, (_Float16)v);
  unsigned int r = (((unsigned int)hb & 0x7fffu) + 0x40u) >> 7;
  r = r > 0x7eu ? 0x7eu : r;
  return (((unsigned int)hb >> 8) & 0x80u) | r;
}

__device__ __forceinline__ float requantf(float v) {
  v *= (1.0f / 256.0f);
  v = fmaxf(v, -128.0f);
  return fminf(v, 127.0f);
}

// load a 32-byte fp8 fragment (4 x 8B chunks at +0,+16,+32,+48) -> v8i
__device__ __forceinline__ v8i load_frag8(const unsigned char* p) {
  union { v8i v; unsigned long long q[4]; } u;
  u.q[0] = *(const unsigned long long*)(p);
  u.q[1] = *(const unsigned long long*)(p + 16);
  u.q[2] = *(const unsigned long long*)(p + 32);
  u.q[3] = *(const unsigned long long*)(p + 48);
  return u.v;
}

// Block-cooperative fp8 GEMM stage:
//   D[256 x (nchunks*64)] = A[256 x 256] * W^T, W fp8 in global (rows = out cols, pitch 256B).
// Weights stream through double-buffered shared LDS chunks (4 ntiles). Each wave
// holds its 32x256 fp8 A panel in 64 VGPRs (two 16-row subtiles), runs two
// interleaved WMMA chains per B fragment, B fragments software-pipelined at
// distance 2 in registers.
template <class Epi>
__device__ __forceinline__ void stage_mm(const unsigned char* __restrict__ Wg,
                                         int nchunks,
                                         const unsigned char* Ab,   // per-wave LDS, 32 x PITB
                                         unsigned char* wbuf,       // shared LDS, 2 x 64 x PITB
                                         Epi&& epi) {
  const int tid  = threadIdx.x;
  const int lane = tid & 31;
  const int r    = lane & 15;
  const int hi8  = (lane >> 4) << 3;           // 0 or 8 bytes

  // A fragments, register-resident for the whole stage (K = 4 steps of 64)
  v8i a0[4], a1[4];
#pragma unroll
  for (int k = 0; k < 4; ++k) {
    a0[k] = load_frag8(Ab + r * PITB + k * 64 + hi8);
    a1[k] = load_frag8(Ab + (16 + r) * PITB + k * 64 + hi8);
  }

  // chunk loader: 64 rows x 256 B = 1024 16B segments over 256 threads
  int lrow[4], lcol[4];
#pragma unroll
  for (int j = 0; j < 4; ++j) {
    int u = tid + j * 256;
    lrow[j] = u >> 4;
    lcol[j] = (u & 15) << 4;
  }
  v4i greg[4];
  auto gload = [&](int c) {
#pragma unroll
    for (int j = 0; j < 4; ++j)
      greg[j] = *(const v4i*)(Wg + (size_t)(c * 64 + lrow[j]) * CH + lcol[j]);
  };
  auto sstore = [&](int p) {
    unsigned char* wb = wbuf + p * WCHUNK_B;
#pragma unroll
    for (int j = 0; j < 4; ++j)
      *(v4i*)(wb + lrow[j] * PITB + lcol[j]) = greg[j];
  };

  gload(0);
  sstore(0);
  __syncthreads();

  for (int c = 0; c < nchunks; ++c) {
    const int p = c & 1;
    if (c + 1 < nchunks) gload(c + 1);         // overlap next chunk's L2 latency
    const unsigned char* wb = wbuf + p * WCHUNK_B;

    auto bfrag = [&](int idx) {                // idx: tt = idx>>2, k = idx&3
      const int tt = idx >> 2, k = idx & 3;
      return load_frag8(wb + (tt * 16 + r) * PITB + k * 64 + hi8);
    };

    v8i bq0 = bfrag(0);
    v8i bq1 = bfrag(1);
    v8f acc0, acc1;
#pragma unroll
    for (int idx = 0; idx < 16; ++idx) {
      const int tt = idx >> 2, k = idx & 3;
      if (k == 0) {
        acc0 = (v8f){0.f, 0.f, 0.f, 0.f, 0.f, 0.f, 0.f, 0.f};
        acc1 = (v8f){0.f, 0.f, 0.f, 0.f, 0.f, 0.f, 0.f, 0.f};
      }
      v8i bcur = (idx & 1) ? bq1 : bq0;
      if (idx + 2 < 16) {                      // prefetch at distance 2
        if (idx & 1) bq1 = bfrag(idx + 2);
        else         bq0 = bfrag(idx + 2);
      }
      acc0 = __builtin_amdgcn_wmma_f32_16x16x64_fp8_fp8(
          a0[k], bcur, (short)0, acc0, false, false);
      acc1 = __builtin_amdgcn_wmma_f32_16x16x64_fp8_fp8(
          a1[k], bcur, (short)0, acc1, false, false);
      if (k == 3) {
        epi(c * 4 + tt, 0, acc0);
        epi(c * 4 + tt, 1, acc1);
      }
    }

    if (c + 1 < nchunks) sstore(1 - p);
    __syncthreads();
  }
}

// ---------- weight conversion kernels ----------
__global__ void cvt_fp8_kernel(const float* __restrict__ src,
                               unsigned char* __restrict__ dst, int n) {
  int i = blockIdx.x * blockDim.x + threadIdx.x;
  if (i < n) dst[i] = (unsigned char)f32_to_e4m3(src[i]);
}

__global__ void cvt_plin_kernel(const float* __restrict__ src,
                                unsigned char* __restrict__ dst) {
  int i = blockIdx.x * blockDim.x + threadIdx.x;
  if (i >= 256 * 256) return;
  int row = i >> 8, col = i & 255;
  dst[i] = (unsigned char)(row < 255 ? f32_to_e4m3(src[row * 256 + col]) : 0u);
}

__global__ void cvt_up1_kernel(const float* __restrict__ src,
                               unsigned char* __restrict__ dstA,
                               float* __restrict__ dstB) {
  int i = blockIdx.x * blockDim.x + threadIdx.x;
  if (i >= 256 * 264) return;
  int row = i / 264, col = i - row * 264;
  float v = src[i];
  if (col < 256) dstA[row * 256 + col] = (unsigned char)f32_to_e4m3(v);
  else           dstB[row * 8 + (col - 256)] = v;
}

// ---------- octree symbol kernel ----------
__global__ void oct_kernel(const int* __restrict__ cur_bin,
                           float* __restrict__ out_oct, int N) {
  int i = blockIdx.x * blockDim.x + threadIdx.x;
  if (i >= N) return;
  int s = 0;
#pragma unroll
  for (int j = 0; j < 8; ++j) s += cur_bin[i * 8 + j] << j;
  out_oct[i] = (float)(short)(s - 1);
}

// ---------- fused main kernel ----------
__global__ __launch_bounds__(WAVES * 32, 1)
void fused_octree_kernel(const float* __restrict__ x,
                         const int* __restrict__ cur_bin,
                         const float* __restrict__ pa_dec,
                         const float* __restrict__ pa_pconv,
                         const float* __restrict__ pa_up1,
                         const float* __restrict__ pa_ur,
                         const char* __restrict__ ws,
                         float* __restrict__ out, int N) {
  extern __shared__ char smem[];
  const int tid  = threadIdx.x;
  const int wave = tid >> 5;
  const int lane = tid & 31;
  const int row0 = blockIdx.x * RPB + wave * MROWS;

  unsigned char* wbuf = (unsigned char*)smem;
  char* slab = smem + LDS_SLAB0 + wave * SLAB_B;
  _Float16*      P0 = (_Float16*)(slab + SLAB_P0);        // x f16, later u1 raw f16
  unsigned char* P1 = (unsigned char*)(slab + SLAB_P1);   // fp8 activations
  unsigned char* P2 = (unsigned char*)(slab + SLAB_P2);   // fp8 activations
  float*       binf = (float*)(slab + SLAB_BIN);

  const unsigned char* wd1  = (const unsigned char*)(ws + OFF_D1);
  const unsigned char* wd2  = (const unsigned char*)(ws + OFF_D2);
  const unsigned char* wpc  = (const unsigned char*)(ws + OFF_PC);
  const unsigned char* wpl  = (const unsigned char*)(ws + OFF_PL);
  const unsigned char* wu1a = (const unsigned char*)(ws + OFF_U1A);
  const unsigned char* wr1  = (const unsigned char*)(ws + OFF_R1);
  const unsigned char* wr2  = (const unsigned char*)(ws + OFF_R2);
  const unsigned char* wuo  = (const unsigned char*)(ws + OFF_UO);
  const float*         u1b  = (const float*)(ws + OFF_U1B);

  const float adec = pa_dec[0], apc = pa_pconv[0];
  const float au1 = pa_up1[0], aur = pa_ur[0];

  // ---- stage x panel: P0 = x (f16), P1 = requant(x) (fp8); bin mask ----
#pragma unroll 4
  for (int i = 0; i < MROWS; ++i) {
    int row = row0 + i;
    int rr  = row < N ? row : N - 1;
    const float* src = x + (size_t)rr * CH + lane * 8;
    v4f v0 = *(const v4f*)(src);
    v4f v1 = *(const v4f*)(src + 4);
    int c = lane * 8;
    v4h r0, r1;
    r0.x = (_Float16)v0.x; r0.y = (_Float16)v0.y; r0.z = (_Float16)v0.z; r0.w = (_Float16)v0.w;
    r1.x = (_Float16)v1.x; r1.y = (_Float16)v1.y; r1.z = (_Float16)v1.z; r1.w = (_Float16)v1.w;
    *(v4h*)(P0 + i * P0PIT + c)     = r0;
    *(v4h*)(P0 + i * P0PIT + c + 4) = r1;
    unsigned long long q = 0;
    q |= (unsigned long long)q8(v0.x);
    q |= (unsigned long long)q8(v0.y) << 8;
    q |= (unsigned long long)q8(v0.z) << 16;
    q |= (unsigned long long)q8(v0.w) << 24;
    q |= (unsigned long long)q8(v1.x) << 32;
    q |= (unsigned long long)q8(v1.y) << 40;
    q |= (unsigned long long)q8(v1.z) << 48;
    q |= (unsigned long long)q8(v1.w) << 56;
    *(unsigned long long*)(P1 + i * PITB + c) = q;
  }
  for (int idx = lane; idx < MROWS * 8; idx += 32) {
    int row = row0 + (idx >> 3);
    binf[idx] = (row < N) ? (float)cur_bin[(size_t)row * 8 + (idx & 7)] : 0.0f;
  }

  const int n  = lane & 15;          // D column owned by this lane
  const int mb = (lane >> 4) << 3;   // D rows mb..mb+7 within a 16-row subtile
  float* pred = out + (size_t)N * 2048;

  // ---- dec resblock ----
  stage_mm(wd1, 4, P1, wbuf, [&](int t, int sub, v8f acc) {
#pragma unroll
    for (int rI = 0; rI < 8; ++rI) {
      float v = acc[rI];
      v = v >= 0.f ? v : adec * v;
      P2[(sub * 16 + mb + rI) * PITB + t * 16 + n] = (unsigned char)q8(v);
    }
  });
  stage_mm(wd2, 4, P2, wbuf, [&](int t, int sub, v8f acc) {
#pragma unroll
    for (int rI = 0; rI < 8; ++rI) {
      int m = sub * 16 + mb + rI, c = t * 16 + n;
      float rec = (float)P0[m * P0PIT + c] + acc[rI];
      P1[m * PITB + c] = (unsigned char)q8(rec);   // recq
    }
  });

  // ---- pred head ----
  stage_mm(wpc, 4, P1, wbuf, [&](int t, int sub, v8f acc) {
#pragma unroll
    for (int rI = 0; rI < 8; ++rI) {
      float v = acc[rI];
      v = v >= 0.f ? v : apc * v;
      P2[(sub * 16 + mb + rI) * PITB + t * 16 + n] = (unsigned char)q8(v);
    }
  });
  stage_mm(wpl, 4, P2, wbuf, [&](int t, int sub, v8f acc) {
    int col = t * 16 + n;
    if (col < 255) {
#pragma unroll
      for (int rI = 0; rI < 8; ++rI) {
        int row = row0 + sub * 16 + mb + rI;
        if (row < N) pred[(size_t)row * 255 + col] = acc[rI];
      }
    }
  });

  // ---- upsample: u1 = prelu(recq @ wu1a^T + binf @ u1b^T) ----
  stage_mm(wu1a, 4, P1, wbuf, [&](int t, int sub, v8f acc) {
    int col = t * 16 + n;
    v4f b0 = *(const v4f*)(u1b + col * 8);
    v4f b1 = *(const v4f*)(u1b + col * 8 + 4);
#pragma unroll
    for (int rI = 0; rI < 8; ++rI) {
      int m = sub * 16 + mb + rI;
      float v = acc[rI];
      v += binf[m * 8 + 0] * b0.x + binf[m * 8 + 1] * b0.y +
           binf[m * 8 + 2] * b0.z + binf[m * 8 + 3] * b0.w;
      v += binf[m * 8 + 4] * b1.x + binf[m * 8 + 5] * b1.y +
           binf[m * 8 + 6] * b1.z + binf[m * 8 + 7] * b1.w;
      v = v >= 0.f ? v : au1 * v;
      P0[m * P0PIT + col] = (_Float16)v;                 // u1 raw (residual)
      P2[m * PITB + col] = (unsigned char)q8(v);         // u1 quantized
    }
  });
  stage_mm(wr1, 4, P2, wbuf, [&](int t, int sub, v8f acc) {
#pragma unroll
    for (int rI = 0; rI < 8; ++rI) {
      float v = acc[rI];
      v = v >= 0.f ? v : aur * v;
      P1[(sub * 16 + mb + rI) * PITB + t * 16 + n] = (unsigned char)q8(v);
    }
  });
  stage_mm(wr2, 4, P1, wbuf, [&](int t, int sub, v8f acc) {
#pragma unroll
    for (int rI = 0; rI < 8; ++rI) {
      int m = sub * 16 + mb + rI, c = t * 16 + n;
      float ures = (float)P0[m * P0PIT + c] + acc[rI];
      P2[m * PITB + c] = (unsigned char)q8(ures);
    }
  });

  // ---- uout: (N,2048), bin-masked ----
  stage_mm(wuo, 32, P2, wbuf, [&](int t, int sub, v8f acc) {
    int col = t * 16 + n;
    int child = col >> 8;
#pragma unroll
    for (int rI = 0; rI < 8; ++rI) {
      int m = sub * 16 + mb + rI;
      int row = row0 + m;
      if (row < N)
        out[(size_t)row * 2048 + col] = acc[rI] * binf[m * 8 + child];
    }
  });
}

// ---------- launch ----------
extern "C" void kernel_launch(void* const* d_in, const int* in_sizes, int n_in,
                              void* d_out, int out_size, void* d_ws, size_t ws_size,
                              hipStream_t stream) {
  const float* x       = (const float*)d_in[0];
  const int*   cur_bin = (const int*)d_in[1];
  const float* w_dec1  = (const float*)d_in[2];
  const float* w_dec2  = (const float*)d_in[3];
  const float* a_dec   = (const float*)d_in[4];
  const float* w_pconv = (const float*)d_in[5];
  const float* a_pconv = (const float*)d_in[6];
  const float* w_plin  = (const float*)d_in[7];
  const float* w_up1   = (const float*)d_in[8];
  const float* a_up1   = (const float*)d_in[9];
  const float* w_ur1   = (const float*)d_in[10];
  const float* w_ur2   = (const float*)d_in[11];
  const float* a_ur    = (const float*)d_in[12];
  const float* w_uout  = (const float*)d_in[13];

  const int N = in_sizes[0] / CH;
  char* ws = (char*)d_ws;

  cvt_fp8_kernel<<<256, 256, 0, stream>>>(w_dec1,  (unsigned char*)(ws + OFF_D1), 65536);
  cvt_fp8_kernel<<<256, 256, 0, stream>>>(w_dec2,  (unsigned char*)(ws + OFF_D2), 65536);
  cvt_fp8_kernel<<<256, 256, 0, stream>>>(w_pconv, (unsigned char*)(ws + OFF_PC), 65536);
  cvt_fp8_kernel<<<256, 256, 0, stream>>>(w_ur1,   (unsigned char*)(ws + OFF_R1), 65536);
  cvt_fp8_kernel<<<256, 256, 0, stream>>>(w_ur2,   (unsigned char*)(ws + OFF_R2), 65536);
  cvt_fp8_kernel<<<2048, 256, 0, stream>>>(w_uout, (unsigned char*)(ws + OFF_UO), 2048 * 256);
  cvt_plin_kernel<<<256, 256, 0, stream>>>(w_plin, (unsigned char*)(ws + OFF_PL));
  cvt_up1_kernel<<<264, 256, 0, stream>>>(w_up1, (unsigned char*)(ws + OFF_U1A),
                                          (float*)(ws + OFF_U1B));

  const int blocks = (N + RPB - 1) / RPB;
  fused_octree_kernel<<<blocks, WAVES * 32, LDS_TOTAL, stream>>>(
      x, cur_bin, a_dec, a_pconv, a_up1, a_ur, ws, (float*)d_out, N);

  float* out_oct = (float*)d_out + (size_t)N * 2048 + (size_t)N * 255;
  oct_kernel<<<(N + 255) / 256, 256, 0, stream>>>(cur_bin, out_oct, N);
}
